// MultiScaleRetention_10737418240743
// MI455X (gfx1250) — matrix operand order repaired and checked by reading, not compiled, not run on hardware
//
#include <hip/hip_runtime.h>
#include <hip/hip_bf16.h>

#define HEADS 8
#define HEAD  64
#define BN    8
#define SS    1024
#define HH    512

typedef __attribute__((ext_vector_type(16))) _Float16 v16h;
typedef __attribute__((ext_vector_type(8)))  _Float16 v8h;
typedef __attribute__((ext_vector_type(8)))  float    v8f;

static __device__ __forceinline__ v8f wmma16(v16h a, v16h b, v8f c) {
    return __builtin_amdgcn_wmma_f32_16x16x32_f16(false, a, false, b, (short)0, c, false, false);
}

// A fragment: 16x32 f16, lane holds row m=lane&15, K chunks {hi8, hi8+16}
static __device__ __forceinline__ v16h load_a(const _Float16* base, int ld) {
    int lane = threadIdx.x & 31;
    const _Float16* p = base + (lane & 15) * ld + ((lane >> 4) << 3);
    v8h lo = *(const v8h*)(p);
    v8h hi = *(const v8h*)(p + 16);
    return __builtin_shufflevector(lo, hi, 0,1,2,3,4,5,6,7,8,9,10,11,12,13,14,15);
}

// B fragment: 32x16 f16, lane holds col n=lane&15, K chunks {hi8, hi8+16}
static __device__ __forceinline__ v16h load_b(const _Float16* base, int ld) {
    int lane = threadIdx.x & 31;
    const _Float16* p = base + (lane & 15) * ld + ((lane >> 4) << 3);
    v8h lo = *(const v8h*)(p);
    v8h hi = *(const v8h*)(p + 16);
    return __builtin_shufflevector(lo, hi, 0,1,2,3,4,5,6,7,8,9,10,11,12,13,14,15);
}

// ---------- stage 0: precision convert / weight transposes ----------
__global__ void k_cvt_x(const float* __restrict__ X, _Float16* __restrict__ Xh, int n) {
    int i = blockIdx.x * 256 + threadIdx.x;
    if (i < n) Xh[i] = (_Float16)X[i];
}
// (G,H,HEAD) -> Wt (G,HEAD,H)
__global__ void k_tr_w(const float* __restrict__ W, _Float16* __restrict__ Wt) {
    int i = blockIdx.x * 256 + threadIdx.x;
    if (i >= HEADS * HH * HEAD) return;
    int g = i / (HH * HEAD);
    int r = i - g * (HH * HEAD);
    int h = r / HEAD, d = r - h * HEAD;
    Wt[(g * HEAD + d) * HH + h] = (_Float16)W[i];
}
// (H,H) -> transpose
__global__ void k_tr_hh(const float* __restrict__ W, _Float16* __restrict__ Wt) {
    int i = blockIdx.x * 256 + threadIdx.x;
    if (i >= HH * HH) return;
    int r = i / HH, c = i - r * HH;
    Wt[c * HH + r] = (_Float16)W[i];
}

// ---------- stage 1: QKV projection + xPos rotary ----------
__global__ void __launch_bounds__(32) k_qkv(const _Float16* __restrict__ Xh,
                      const _Float16* __restrict__ WQt, const _Float16* __restrict__ WKt,
                      const _Float16* __restrict__ WVt,
                      _Float16* __restrict__ Qh, _Float16* __restrict__ Kh,
                      _Float16* __restrict__ Vt) {
    int blk = blockIdx.x;
    int st = blk & 63; blk >>= 6;
    int bn = blk & 7;  int g = blk >> 3;
    int s0 = st * 16;
    int lane = threadIdx.x;
    int n = lane & 15, hi8 = (lane >> 4) << 3;

    v8f qa[4], ka[4], va[4];
#pragma unroll
    for (int t = 0; t < 4; ++t) { qa[t] = (v8f)(0.f); ka[t] = (v8f)(0.f); va[t] = (v8f)(0.f); }

    const _Float16* xbase = Xh + (bn * SS + s0) * HH;
    for (int kk = 0; kk < 16; ++kk) {
        v16h af = load_a(xbase + kk * 32, HH);
        // batch Q and K fragment loads (8 in flight) before consuming -> partial loadcnt waits
        v16h bq[4], bk[4], bv[4];
#pragma unroll
        for (int nt = 0; nt < 4; ++nt) {
            int wrow = (g * HEAD + nt * 16) * HH + kk * 32;
            bq[nt] = load_b(WQt + wrow, HH);
            bk[nt] = load_b(WKt + wrow, HH);
        }
#pragma unroll
        for (int nt = 0; nt < 4; ++nt) {
            int wrow = (g * HEAD + nt * 16) * HH + kk * 32;
            bv[nt] = load_b(WVt + wrow, HH);
            qa[nt] = wmma16(af, bq[nt], qa[nt]);
            ka[nt] = wmma16(af, bk[nt], ka[nt]);
        }
#pragma unroll
        for (int nt = 0; nt < 4; ++nt)
            va[nt] = wmma16(af, bv[nt], va[nt]);
    }

    int gbn = g * BN + bn;
#pragma unroll
    for (int nt = 0; nt < 4; ++nt) {
        int d = nt * 16 + n;
        int i = d >> 1;
        float invf = __expf(-(float)i * (9.2103403720f / 32.f)); // 10000^(-i/32)
        float lsv  = __logf((2.f * (float)i + 25.6f) / 89.6f);   // ln scale_vec
        v8h vpack;
#pragma unroll
        for (int r = 0; r < 8; ++r) {
            int s = s0 + hi8 + r;
            float ang = (float)s * invf;
            float sn, cs;
            __sincosf(ang, &sn, &cs);
            float scl = __expf(((float)s / 512.f) * lsv);
            float q = qa[nt][r], k = ka[nt][r];
            float qp = __shfl_xor(q, 1, 32);
            float kp = __shfl_xor(k, 1, 32);
            float qr = (d & 1) ? qp : -qp;
            float kr = (d & 1) ? kp : -kp;
            float qo = (q * cs + qr * sn) * scl;
            float ko = (k * cs + kr * sn) / scl;
            int row = gbn * SS + s;
            Qh[row * HEAD + d] = (_Float16)qo;
            Kh[row * HEAD + d] = (_Float16)ko;
            vpack[r] = (_Float16)va[nt][r];
        }
        *(v8h*)(Vt + (gbn * HEAD + d) * SS + s0 + hi8) = vpack;
    }
}

// ---------- stage 2: retention (scores*decay)@V + GroupNorm ----------
__global__ void __launch_bounds__(32) k_attn(const _Float16* __restrict__ Qh,
                       const _Float16* __restrict__ Kh, const _Float16* __restrict__ Vt,
                       const float* __restrict__ gnw, const float* __restrict__ gnb,
                       _Float16* __restrict__ Ycat) {
    __shared__ __align__(16) _Float16 sS[16 * 32];
    int blk = blockIdx.x;
    int st = blk & 63; blk >>= 6;
    int bn = blk & 7;  int g = blk >> 3;
    int lane = threadIdx.x;
    int n = lane & 15, hi8 = (lane >> 4) << 3;
    int s0 = st * 16;
    int gbn = g * BN + bn;

    const float la = -3.4657359028f;   // ln(1/32)
    const float lb = -6.2383246250f;   // ln(1/512)
    float gamma = 1.f - __expf(la + (lb - la) * ((float)g / 7.f));
    float lgam = __logf(gamma);

    const _Float16* qbase = Qh + (gbn * SS + s0) * HEAD;
    v16h q0 = load_a(qbase, HEAD);
    v16h q1 = load_a(qbase + 32, HEAD);

    v8f y[4];
#pragma unroll
    for (int t = 0; t < 4; ++t) y[t] = (v8f)(0.f);

    int npair = (st >> 1) + 1;
    for (int tp = 0; tp < npair; ++tp) {
        int kb = tp * 32;
        if (tp + 1 < npair) {
            __builtin_prefetch(Kh + (gbn * SS + kb + 32) * HEAD, 0, 1);
            __builtin_prefetch(Vt + (gbn * HEAD) * SS + kb + 32, 0, 1);
        }
        // batch the 4 K-fragments for both 16-col score tiles
        v16h kf[4];
        const _Float16* kbase0 = Kh + (gbn * SS + kb) * HEAD;
        const _Float16* kbase1 = Kh + (gbn * SS + kb + 16) * HEAD;
        kf[0] = load_b(kbase0, HEAD);
        kf[1] = load_b(kbase0 + 32, HEAD);
        kf[2] = load_b(kbase1, HEAD);
        kf[3] = load_b(kbase1 + 32, HEAD);

        v8f sc0 = (v8f)(0.f), sc1 = (v8f)(0.f);
        sc0 = wmma16(q0, kf[0], sc0);
        sc0 = wmma16(q1, kf[1], sc0);
        sc1 = wmma16(q0, kf[2], sc1);
        sc1 = wmma16(q1, kf[3], sc1);

        // hoist V fragment loads so they overlap the LDS round-trip
        v16h vf[4];
#pragma unroll
        for (int nt = 0; nt < 4; ++nt)
            vf[nt] = load_b(Vt + (gbn * HEAD + nt * 16) * SS + kb, SS);

        // decay mask + C-layout -> A-layout transpose through LDS
#pragma unroll
        for (int r = 0; r < 8; ++r) {
            int sidx = s0 + hi8 + r;
            int d0 = sidx - (kb + n);
            int d1 = sidx - (kb + 16 + n);
            float v0 = (d0 >= 0) ? sc0[r] * __expf((float)d0 * lgam) : 0.f;
            float v1 = (d1 >= 0) ? sc1[r] * __expf((float)d1 * lgam) : 0.f;
            sS[(hi8 + r) * 32 + n]      = (_Float16)v0;
            sS[(hi8 + r) * 32 + 16 + n] = (_Float16)v1;
        }
        asm volatile("s_wait_dscnt 0" ::: "memory");
        v8h slo = *(const v8h*)&sS[(lane & 15) * 32 + hi8];
        v8h shi = *(const v8h*)&sS[(lane & 15) * 32 + hi8 + 16];
        v16h sf = __builtin_shufflevector(slo, shi, 0,1,2,3,4,5,6,7,8,9,10,11,12,13,14,15);
#pragma unroll
        for (int nt = 0; nt < 4; ++nt)
            y[nt] = wmma16(sf, vf[nt], y[nt]);
    }

    // GroupNorm over d=64 per row, then scatter into Ycat (f16)
#pragma unroll
    for (int r = 0; r < 8; ++r) {
        float s1 = y[0][r] + y[1][r] + y[2][r] + y[3][r];
        float s2 = y[0][r]*y[0][r] + y[1][r]*y[1][r] + y[2][r]*y[2][r] + y[3][r]*y[3][r];
#pragma unroll
        for (int m = 1; m < 16; m <<= 1) {
            s1 += __shfl_xor(s1, m, 32);
            s2 += __shfl_xor(s2, m, 32);
        }
        float mean = s1 * (1.f / 64.f);
        float var  = s2 * (1.f / 64.f) - mean * mean;
        float inv  = rsqrtf(var + 1e-5f);
        int row = bn * SS + s0 + hi8 + r;
#pragma unroll
        for (int nt = 0; nt < 4; ++nt) {
            int d = g * HEAD + nt * 16 + n;
            float val = (y[nt][r] - mean) * inv * gnw[d] + gnb[d];
            Ycat[row * HH + d] = (_Float16)val;
        }
    }
}

// ---------- stage 3a: gate = silu(X@W_G) * Ycat ----------
__global__ void __launch_bounds__(32) k_gate(const _Float16* __restrict__ Xh,
                       const _Float16* __restrict__ WGt,
                       const _Float16* __restrict__ Ycat, _Float16* __restrict__ Gh) {
    int blk = blockIdx.x;
    int hb = blk & 7;  blk >>= 3;
    int st = blk & 63; blk >>= 6;
    int bn = blk;
    int s0 = st * 16;
    int lane = threadIdx.x;
    int n = lane & 15, hi8 = (lane >> 4) << 3;

    v8f acc[4];
#pragma unroll
    for (int t = 0; t < 4; ++t) acc[t] = (v8f)(0.f);

    const _Float16* xb = Xh + (bn * SS + s0) * HH;
    for (int kk = 0; kk < 16; ++kk) {
        v16h af = load_a(xb + kk * 32, HH);
        v16h bf[4];
#pragma unroll
        for (int nt = 0; nt < 4; ++nt)
            bf[nt] = load_b(WGt + (hb * 64 + nt * 16) * HH + kk * 32, HH);
#pragma unroll
        for (int nt = 0; nt < 4; ++nt)
            acc[nt] = wmma16(af, bf[nt], acc[nt]);
    }
#pragma unroll
    for (int nt = 0; nt < 4; ++nt) {
#pragma unroll
        for (int r = 0; r < 8; ++r) {
            int col = hb * 64 + nt * 16 + n;
            int row = bn * SS + s0 + hi8 + r;
            float x = acc[nt][r];
            float gate = x / (1.f + __expf(-x));      // x * sigmoid(x)
            float yv = (float)Ycat[row * HH + col];
            Gh[row * HH + col] = (_Float16)(gate * yv);
        }
    }
}

// ---------- stage 3b: out = G @ W_O (f32 out) ----------
__global__ void __launch_bounds__(32) k_out(const _Float16* __restrict__ Gh,
                      const _Float16* __restrict__ WOt, float* __restrict__ out) {
    int blk = blockIdx.x;
    int hb = blk & 7;  blk >>= 3;
    int st = blk & 63; blk >>= 6;
    int bn = blk;
    int s0 = st * 16;
    int lane = threadIdx.x;
    int n = lane & 15, hi8 = (lane >> 4) << 3;

    v8f acc[4];
#pragma unroll
    for (int t = 0; t < 4; ++t) acc[t] = (v8f)(0.f);

    const _Float16* gb = Gh + (bn * SS + s0) * HH;
    for (int kk = 0; kk < 16; ++kk) {
        v16h af = load_a(gb + kk * 32, HH);
        v16h bf[4];
#pragma unroll
        for (int nt = 0; nt < 4; ++nt)
            bf[nt] = load_b(WOt + (hb * 64 + nt * 16) * HH + kk * 32, HH);
#pragma unroll
        for (int nt = 0; nt < 4; ++nt)
            acc[nt] = wmma16(af, bf[nt], acc[nt]);
    }
#pragma unroll
    for (int nt = 0; nt < 4; ++nt) {
#pragma unroll
        for (int r = 0; r < 8; ++r) {
            int col = hb * 64 + nt * 16 + n;
            int row = bn * SS + s0 + hi8 + r;
            out[row * HH + col] = acc[nt][r];
        }
    }
}

extern "C" void kernel_launch(void* const* d_in, const int* in_sizes, int n_in,
                              void* d_out, int out_size, void* d_ws, size_t ws_size,
                              hipStream_t stream) {
    const float* X   = (const float*)d_in[0];
    const float* W_Q = (const float*)d_in[1];
    const float* W_K = (const float*)d_in[2];
    const float* W_V = (const float*)d_in[3];
    const float* W_G = (const float*)d_in[4];
    const float* W_O = (const float*)d_in[5];
    const float* gnw = (const float*)d_in[6];
    const float* gnb = (const float*)d_in[7];
    float* out = (float*)d_out;

    size_t off = 0;
    auto alloc = [&](size_t bytes) {
        void* p = (char*)d_ws + off;
        off += (bytes + 255) & ~(size_t)255;
        return p;
    };
    _Float16* Xh   = (_Float16*)alloc((size_t)BN * SS * HH * 2);
    _Float16* WQt  = (_Float16*)alloc((size_t)HEADS * HEAD * HH * 2);
    _Float16* WKt  = (_Float16*)alloc((size_t)HEADS * HEAD * HH * 2);
    _Float16* WVt  = (_Float16*)alloc((size_t)HEADS * HEAD * HH * 2);
    _Float16* WGt  = (_Float16*)alloc((size_t)HH * HH * 2);
    _Float16* WOt  = (_Float16*)alloc((size_t)HH * HH * 2);
    _Float16* Qh   = (_Float16*)alloc((size_t)HEADS * BN * SS * HEAD * 2);
    _Float16* Kh   = (_Float16*)alloc((size_t)HEADS * BN * SS * HEAD * 2);
    _Float16* Vt   = (_Float16*)alloc((size_t)HEADS * BN * HEAD * SS * 2);
    _Float16* Ycat = (_Float16*)alloc((size_t)BN * SS * HH * 2);
    _Float16* Gh   = (_Float16*)alloc((size_t)BN * SS * HH * 2);
    (void)ws_size; (void)in_sizes; (void)n_in; (void)out_size;

    int nx = BN * SS * HH;
    k_cvt_x<<<(nx + 255) / 256, 256, 0, stream>>>(X, Xh, nx);
    int nw = HEADS * HH * HEAD;
    k_tr_w<<<(nw + 255) / 256, 256, 0, stream>>>(W_Q, WQt);
    k_tr_w<<<(nw + 255) / 256, 256, 0, stream>>>(W_K, WKt);
    k_tr_w<<<(nw + 255) / 256, 256, 0, stream>>>(W_V, WVt);
    int nh = HH * HH;
    k_tr_hh<<<(nh + 255) / 256, 256, 0, stream>>>(W_G, WGt);
    k_tr_hh<<<(nh + 255) / 256, 256, 0, stream>>>(W_O, WOt);

    k_qkv<<<HEADS * BN * (SS / 16), 32, 0, stream>>>(Xh, WQt, WKt, WVt, Qh, Kh, Vt);
    k_attn<<<HEADS * BN * (SS / 16), 32, 0, stream>>>(Qh, Kh, Vt, gnw, gnb, Ycat);
    k_gate<<<BN * (SS / 16) * (HH / 64), 32, 0, stream>>>(Xh, WGt, Ycat, Gh);
    k_out<<<BN * (SS / 16) * (HH / 64), 32, 0, stream>>>(Gh, WOt, out);
}
